// BlockDiagonalLRU_11433202942760
// MI455X (gfx1250) — matrix operand order repaired and checked by reading, not compiled, not run on hardware
//
#include <hip/hip_runtime.h>
#include <stdint.h>

// ---------- problem constants ----------
#define Dd   1024
#define Mm   16
#define Hh   64          // D / M
#define Bb   4
#define Tt   1024
#define NTOK (Bb * Tt)   // 4096
#define NA   (Hh * Mm * (Mm + 1))  // 17408

// ---------- WMMA vector types ----------
typedef __attribute__((ext_vector_type(16))) __bf16 v16bf;
typedef __attribute__((ext_vector_type(8)))  __bf16 v8bf;
typedef __attribute__((ext_vector_type(8)))  float  v8f;

// ---------- bf16 helpers (bit-level, no __bf16 arithmetic) ----------
__device__ __forceinline__ unsigned short f32_to_bf16(float f) {
    union { float f; unsigned u; } v; v.f = f;
    unsigned r = v.u + 0x7FFFu + ((v.u >> 16) & 1u);   // round-to-nearest-even
    return (unsigned short)(r >> 16);
}
__device__ __forceinline__ float bf16_to_f32(unsigned short h) {
    union { float f; unsigned u; } v; v.u = ((unsigned)h) << 16;
    return v.f;
}

// ---------- WMMA fragment loaders (bf16 storage as unsigned short) ----------
// A-matrix 16x32 bf16, row-major source (lda in elements).
// lanes 0-15: row = lane,  K chunks [k0..k0+7] (v0-3) and [k0+16..k0+23] (v4-7)
// lanes 16-31: row = lane-16, chunks at +8
__device__ __forceinline__ v16bf load_a_frag(const unsigned short* A, int lda,
                                             int row0, int k0) {
    int lane = threadIdx.x & 31;
    int r  = row0 + (lane & 15);
    int kb = k0 + ((lane >> 4) << 3);
    const __bf16* p = (const __bf16*)(A + (size_t)r * lda + kb);
    v8bf lo = *(const v8bf*)p;          // 16B
    v8bf hi = *(const v8bf*)(p + 16);   // 16B
    v16bf f;
#pragma unroll
    for (int i = 0; i < 8; ++i) { f[i] = lo[i]; f[i + 8] = hi[i]; }
    return f;
}

// B-matrix 32x16 bf16 from row-major weights W(N x K): B[k][n] = W[col0+n][k0+k].
// lanes 0-15: col = lane, K chunks [k0..k0+7] and [k0+16..k0+23]; lanes 16-31: chunks at +8
__device__ __forceinline__ v16bf load_b_frag(const unsigned short* W, int ldb,
                                             int col0, int k0) {
    int lane = threadIdx.x & 31;
    int c  = col0 + (lane & 15);
    int kb = k0 + ((lane >> 4) << 3);
    const __bf16* p = (const __bf16*)(W + (size_t)c * ldb + kb);
    v8bf lo = *(const v8bf*)p;
    v8bf hi = *(const v8bf*)(p + 16);
    v16bf f;
#pragma unroll
    for (int i = 0; i < 8; ++i) { f[i] = lo[i]; f[i + 8] = hi[i]; }
    return f;
}

#define WMMA_BF16(a, b, c) \
    __builtin_amdgcn_wmma_f32_16x16x32_bf16(false, (a), false, (b), (short)0, (c), false, false)

// ---------- kernel 1: f32 -> bf16 convert (grid-stride) ----------
__global__ void __launch_bounds__(256)
cvt_bf16_kernel(const float* __restrict__ in, unsigned short* __restrict__ out, int n) {
    for (int i = blockIdx.x * 256 + threadIdx.x; i < n; i += gridDim.x * 256)
        out[i] = f32_to_bf16(in[i]);
}

// ---------- kernel 2: RMSNorm -> bf16 h ----------
__global__ void __launch_bounds__(256)
rmsnorm_kernel(const float* __restrict__ x, const float* __restrict__ w,
               unsigned short* __restrict__ hbf) {
    int token = blockIdx.x;                 // 0..4095
    const float* xr = x + (size_t)token * Dd;
    float vals[4];
    float ss = 0.f;
#pragma unroll
    for (int i = 0; i < 4; ++i) {
        vals[i] = xr[threadIdx.x + i * 256];
        ss += vals[i] * vals[i];
    }
    __shared__ float red[256];
    red[threadIdx.x] = ss;
    __syncthreads();
#pragma unroll
    for (int s = 128; s > 0; s >>= 1) {
        if (threadIdx.x < s) red[threadIdx.x] += red[threadIdx.x + s];
        __syncthreads();
    }
    float scale = rsqrtf(red[0] / (float)Dd + 1e-5f);
#pragma unroll
    for (int i = 0; i < 4; ++i) {
        int d = threadIdx.x + i * 256;
        hbf[(size_t)token * Dd + d] = f32_to_bf16(vals[i] * scale * w[d]);
    }
}

// ---------- kernel 3: register-blocked bf16 WMMA GEMM ----------
// C = A(4096xK) * W(NxK)^T (+res). Wave computes 32x32 (2x2 WMMA tiles):
// 2 A-frags + 2 B-frags feed 4 WMMAs -> load:wmma = 2:1, halving L2 operand
// traffic. Workgroup = 8 waves (2x4) -> 64x128 block tile. No LDS staging:
// operands are resident in MI455X's 192MB L2.
__global__ void __launch_bounds__(256)
gemm_bf16_wmma(const unsigned short* __restrict__ A, const unsigned short* __restrict__ W,
               const float* __restrict__ residual, float* __restrict__ C,
               int N, int K) {
    int wave = threadIdx.x >> 5;
    int lane = threadIdx.x & 31;
    int r0 = blockIdx.y * 64  + (wave >> 2) * 32;
    int c0 = blockIdx.x * 128 + (wave & 3) * 32;

    v8f acc[2][2] = {};
    for (int k0 = 0; k0 < K; k0 += 32) {
        v16bf af0 = load_a_frag(A, K, r0,      k0);
        v16bf af1 = load_a_frag(A, K, r0 + 16, k0);
        v16bf bf0 = load_b_frag(W, K, c0,      k0);
        v16bf bf1 = load_b_frag(W, K, c0 + 16, k0);
        acc[0][0] = WMMA_BF16(af0, bf0, acc[0][0]);
        acc[0][1] = WMMA_BF16(af0, bf1, acc[0][1]);
        acc[1][0] = WMMA_BF16(af1, bf0, acc[1][0]);
        acc[1][1] = WMMA_BF16(af1, bf1, acc[1][1]);
    }
    // C/D layout: VGPR g -> M = g (lanes 0-15) / 8+g (lanes 16-31); N = lane&15
    int rbase = (lane >> 4) * 8;
    int cbase = lane & 15;
#pragma unroll
    for (int ri = 0; ri < 2; ++ri) {
#pragma unroll
        for (int ci = 0; ci < 2; ++ci) {
            int row = r0 + ri * 16 + rbase;
            int col = c0 + ci * 16 + cbase;
#pragma unroll
            for (int g = 0; g < 8; ++g) {
                size_t idx = (size_t)(row + g) * N + col;
                float o = acc[ri][ci][g];
                if (residual) o += residual[idx];
                C[idx] = o;
            }
        }
    }
}

// ---------- kernel 4: fused W_a GEMM + softmax over 17 ----------
// Block: 32 tokens x one h (272 cols = 17 WMMA col tiles). 8 waves: wave w owns
// col tiles {w, w+8, (16 if w==0)}; per k-step it loads 2 A-row frags and reuses
// each B-frag for 2 WMMAs (register reuse of the largest operand stream).
// Epilogue: logits -> LDS -> per-(token,m) softmax -> a0 (f32) + A (bf16),
// both in scan-friendly [b][h][t][...] layout.
__global__ void __launch_bounds__(256)
gates_kernel(const unsigned short* __restrict__ hbf, const unsigned short* __restrict__ wabf,
             float* __restrict__ a0_ws, unsigned short* __restrict__ A_ws) {
    int tok0 = blockIdx.x * 32;   // global token tile (32 tokens)
    int h    = blockIdx.y;        // 0..63
    int wave = threadIdx.x >> 5;
    int lane = threadIdx.x & 31;

    __shared__ float logits[32][272];   // 34816 B

    v8f acc[3][2];
    int tiles[3];
    int nt = 0;
    for (int tile = wave; tile < 17; tile += 8) {
        tiles[nt] = tile;
        v8f z = {};
        acc[nt][0] = z;
        acc[nt][1] = z;
        ++nt;
    }

    for (int k0 = 0; k0 < Dd; k0 += 32) {
        v16bf af0 = load_a_frag(hbf, Dd, tok0,      k0);
        v16bf af1 = load_a_frag(hbf, Dd, tok0 + 16, k0);
        for (int i = 0; i < nt; ++i) {
            int c0 = h * 272 + tiles[i] * 16;
            v16bf bf = load_b_frag(wabf, Dd, c0, k0);
            acc[i][0] = WMMA_BF16(af0, bf, acc[i][0]);
            acc[i][1] = WMMA_BF16(af1, bf, acc[i][1]);
        }
    }

    int rb = (lane >> 4) * 8;
    for (int i = 0; i < nt; ++i) {
        int col = tiles[i] * 16 + (lane & 15);
#pragma unroll
        for (int rg = 0; rg < 2; ++rg)
#pragma unroll
            for (int g = 0; g < 8; ++g)
                logits[rg * 16 + rb + g][col] = acc[i][rg][g];
    }
    __syncthreads();

    // softmax: 32 tokens x 16 m = 512 rows of 17; 256 threads -> 2 each
#pragma unroll
    for (int it = 0; it < 2; ++it) {
        int idx   = threadIdx.x + it * 256;
        int token = idx >> 4;
        int m     = idx & 15;
        float l[17];
        float mx = -1e30f;
#pragma unroll
        for (int e = 0; e < 17; ++e) {
            l[e] = logits[token][m * 17 + e];
            mx = fmaxf(mx, l[e]);
        }
        float s = 0.f;
#pragma unroll
        for (int e = 0; e < 17; ++e) { l[e] = __expf(l[e] - mx); s += l[e]; }
        float inv = 1.f / s;

        int n = tok0 + token;
        int b = n >> 10;          // T = 1024
        int t = n & 1023;
        size_t base = ((size_t)(b * Hh + h) * Tt + t);
        a0_ws[base * Mm + m] = l[0] * inv;

        // pack 16 bf16 A-row entries into 2x uint4 (32B aligned store)
        unsigned pk[8];
#pragma unroll
        for (int j = 0; j < 8; ++j) {
            unsigned lo = f32_to_bf16(l[2 * j + 1] * inv);
            unsigned hi = f32_to_bf16(l[2 * j + 2] * inv);
            pk[j] = lo | (hi << 16);
        }
        unsigned short* arow = A_ws + base * (Mm * Mm) + m * Mm;
        ((uint4*)arow)[0] = make_uint4(pk[0], pk[1], pk[2], pk[3]);
        ((uint4*)arow)[1] = make_uint4(pk[4], pk[5], pk[6], pk[7]);
    }
}

// ---------- kernel 5: sequential linear scan over T ----------
// One wave per (b,h). Lane i owns state[i]; per step: coalesced 512B bf16 A-row
// read, LDS state broadcast, 16 FMAs. Prefetch a few steps ahead.
__global__ void __launch_bounds__(32)
scan_kernel(const unsigned short* __restrict__ A_ws, const float* __restrict__ a0_ws,
            const float* __restrict__ v, unsigned short* __restrict__ hout) {
    int bh = blockIdx.x;          // 0..255
    int b  = bh / Hh;
    int h  = bh % Hh;
    int lane = threadIdx.x;

    __shared__ float s[16];
    float state = 0.f;

    const unsigned short* Abase = A_ws + (size_t)bh * Tt * (Mm * Mm);
    const float*          a0b   = a0_ws + (size_t)bh * Tt * Mm;

    for (int t = 0; t < Tt; ++t) {
        if (lane < 16) s[lane] = state;
        __syncthreads();

        float nv = 0.f;
        if (lane < 16) {
            const unsigned short* arow = Abase + (size_t)t * 256 + lane * 16;
            if (t + 4 < Tt)
                __builtin_prefetch(arow + 4 * 256, 0, 1);   // global_prefetch_b8
            uint4 q0 = ((const uint4*)arow)[0];
            uint4 q1 = ((const uint4*)arow)[1];
            unsigned qs[8] = {q0.x, q0.y, q0.z, q0.w, q1.x, q1.y, q1.z, q1.w};
            float a[16];
#pragma unroll
            for (int j = 0; j < 8; ++j) {
                a[2 * j]     = bf16_to_f32((unsigned short)(qs[j] & 0xffffu));
                a[2 * j + 1] = bf16_to_f32((unsigned short)(qs[j] >> 16));
            }
#pragma unroll
            for (int j = 0; j < 16; ++j) nv += a[j] * s[j];
            float u = a0b[t * Mm + lane] *
                      v[((size_t)b * Tt + t) * Dd + h * Mm + lane];
            nv += u;
        }
        __syncthreads();
        state = nv;
        if (lane < 16)
            hout[((size_t)b * Tt + t) * Dd + h * Mm + lane] = f32_to_bf16(state);
    }
}

// ---------- host launcher ----------
extern "C" void kernel_launch(void* const* d_in, const int* in_sizes, int n_in,
                              void* d_out, int out_size, void* d_ws, size_t ws_size,
                              hipStream_t stream) {
    const float* x      = (const float*)d_in[0];   // (B,T,D)
    const float* norm_w = (const float*)d_in[1];   // (D)
    const float* W_v    = (const float*)d_in[2];   // (D,D)
    const float* W_a    = (const float*)d_in[3];   // (NA,D)
    const float* W_out  = (const float*)d_in[4];   // (D,D)
    float* out = (float*)d_out;

    // workspace carve-up (bf16 stored as unsigned short)
    char* ws = (char*)d_ws;
    size_t off = 0;
    auto alloc = [&](size_t bytes) {
        void* p = ws + off;
        off += (bytes + 255) & ~(size_t)255;
        return p;
    };
    unsigned short* hbf   = (unsigned short*)alloc((size_t)NTOK * Dd * 2);   //  8 MB
    unsigned short* wvbf  = (unsigned short*)alloc((size_t)Dd * Dd * 2);     //  2 MB
    unsigned short* wabf  = (unsigned short*)alloc((size_t)NA * Dd * 2);     // 36 MB
    unsigned short* wobf  = (unsigned short*)alloc((size_t)Dd * Dd * 2);     //  2 MB
    float*          vbuf  = (float*)alloc((size_t)NTOK * Dd * 4);            // 16 MB
    float*          a0ws  = (float*)alloc((size_t)Bb * Hh * Tt * Mm * 4);    // 16 MB
    unsigned short* Aws   = (unsigned short*)alloc((size_t)Bb * Hh * Tt * Mm * Mm * 2); // 134 MB
    unsigned short* hobf  = (unsigned short*)alloc((size_t)NTOK * Dd * 2);   //  8 MB
    (void)ws_size; (void)n_in; (void)in_sizes; (void)out_size;

    // 1) weight converts f32 -> bf16
    cvt_bf16_kernel<<<512, 256, 0, stream>>>(W_v,   wvbf, Dd * Dd);
    cvt_bf16_kernel<<<2048, 256, 0, stream>>>(W_a,  wabf, NA * Dd);
    cvt_bf16_kernel<<<512, 256, 0, stream>>>(W_out, wobf, Dd * Dd);

    // 2) RMSNorm -> h (bf16)
    rmsnorm_kernel<<<NTOK, 256, 0, stream>>>(x, norm_w, hbf);

    // 3) v = h @ W_v^T   (f32 out)
    gemm_bf16_wmma<<<dim3(Dd / 128, NTOK / 64), 256, 0, stream>>>(
        hbf, wvbf, nullptr, vbuf, Dd, Dd);

    // 4) gate logits + softmax -> a0, A
    gates_kernel<<<dim3(NTOK / 32, Hh), 256, 0, stream>>>(hbf, wabf, a0ws, Aws);

    // 5) sequential scan over T
    scan_kernel<<<Bb * Hh, 32, 0, stream>>>(Aws, a0ws, vbuf, hobf);

    // 6) out = x + h_out @ W_out^T
    gemm_bf16_wmma<<<dim3(Dd / 128, NTOK / 64), 256, 0, stream>>>(
        hobf, wobf, x, out, Dd, Dd);
}
